// PointMOS_23519240913348
// MI455X (gfx1250) — matrix operand order repaired and checked by reading, not compiled, not run on hardware
//
#include <hip/hip_runtime.h>
#include <hip/hip_bf16.h>
#include <math.h>

#define NPTS 32768

typedef __attribute__((ext_vector_type(16))) _Float16 v16h;
typedef __attribute__((ext_vector_type(8)))  _Float16 v8h;
typedef __attribute__((ext_vector_type(4)))  _Float16 v4h;
typedef __attribute__((ext_vector_type(8)))  float    v8f;

// ---------------------------------------------------------------------------
// GEMM: Y[M,N] = relu((X[M,Kp] @ W^T + bias) * gain + beta)
//  - X row stride Kp (multiple of 32, zero-padded), M multiple of 64
//  - Wp pre-packed in B-fragment order: [ntile][kchunk][lane][16 f16]
//  - grid.x = M/64, grid.y = N/64, block = 128 (4 waves)
//  - each wave: 16 rows x 64 cols (4 WMMA n-tiles), f32 accumulate
// ---------------------------------------------------------------------------
__global__ void k_gemm(const float* __restrict__ X, const v16h* __restrict__ Wp,
                       const float* __restrict__ bias, const float* __restrict__ gain,
                       const float* __restrict__ beta, float* __restrict__ Y,
                       int M, int Kp, int N)
{
    __shared__ _Float16 sX[64][40];          // 64 rows x 32 k, stride 40 (80B, conflict-free)
    const int tid  = threadIdx.x;
    const int wave = tid >> 5, lane = tid & 31;
    const int half = lane >> 4, lr = lane & 15;
    const int m0  = blockIdx.x * 64;
    const int nt0 = blockIdx.y * 4;
    const int KC  = Kp >> 5;

    v8f acc[4] = {};
    for (int kc = 0; kc < KC; ++kc) {
        // stage 64x32 f32 tile -> f16 LDS (vector loads, no guards)
        #pragma unroll
        for (int t = tid; t < 512; t += 128) {
            int r = t >> 3, c4 = t & 7;
            const float4 vv = *(const float4*)(X + (size_t)(m0 + r) * Kp + (kc << 5) + (c4 << 2));
            v4h h; h[0] = (_Float16)vv.x; h[1] = (_Float16)vv.y;
                   h[2] = (_Float16)vv.z; h[3] = (_Float16)vv.w;
            *(v4h*)&sX[r][c4 << 2] = h;
        }
        __syncthreads();

        // A fragment: two 16B LDS reads per lane (ISA 16-bit A 16x32 layout)
        v8h alo = *(const v8h*)&sX[wave * 16 + lr][half * 8];
        v8h ahi = *(const v8h*)&sX[wave * 16 + lr][16 + half * 8];
        v16h afrag = __builtin_shufflevector(alo, ahi, 0,1,2,3,4,5,6,7,8,9,10,11,12,13,14,15);

        // B fragments: one contiguous 32B load per lane per tile
        const v16h* wb = Wp + (size_t)kc * 32 + lane;
        #pragma unroll
        for (int j = 0; j < 4; ++j) {
            v16h bfrag = wb[(size_t)(nt0 + j) * KC * 32];
            acc[j] = __builtin_amdgcn_wmma_f32_16x16x32_f16(false, afrag, false, bfrag,
                                                            (short)0, acc[j], false, false);
        }
        __syncthreads();
    }

    // epilogue: lane column n, rows wave*16 + r + 8*half
    #pragma unroll
    for (int j = 0; j < 4; ++j) {
        int n = (nt0 + j) * 16 + lr;
        float bb = bias[n], gg = gain[n], be = beta[n];
        #pragma unroll
        for (int r = 0; r < 8; ++r) {
            int m = m0 + wave * 16 + r + half * 8;
            float v = (acc[j][r] + bb) * gg + be;
            Y[(size_t)m * N + n] = fmaxf(v, 0.f);
        }
    }
}

// Pack W (N,K) f32 -> fragment-ordered f16: flat idx = ((nt*KC+kc)*32+lane)*16+e
__global__ void k_pack(const float* __restrict__ W, _Float16* __restrict__ Wp,
                       int K, int Kp, int N)
{
    int t = blockIdx.x * blockDim.x + threadIdx.x;
    if (t >= N * Kp) return;
    int KC   = Kp >> 5;
    int e    = t & 15;
    int lane = (t >> 4) & 31;
    int kc   = (t >> 9) % KC;
    int nt   = (t >> 9) / KC;
    int n = (nt << 4) + (lane & 15);
    int k = (kc << 5) + ((e >> 3) << 4) + ((lane >> 4) << 3) + (e & 7);
    Wp[t] = (k < K) ? (_Float16)W[(size_t)n * K + k] : (_Float16)0.f;
}

// ---------------------------------------------------------------------------
// Preprocess: enh features + vertex attention (only w[:,5] needed) -> x0, xyz
// ---------------------------------------------------------------------------
__global__ void k_preprocess(const float* __restrict__ pts,
                             const float* __restrict__ W1, const float* __restrict__ b1,
                             const float* __restrict__ g1, const float* __restrict__ be1,
                             const float* __restrict__ W2, const float* __restrict__ b2,
                             float* __restrict__ xyz, float* __restrict__ x0)
{
    int i = blockIdx.x * blockDim.x + threadIdx.x;
    if (i >= NPTS) return;
    const float* p = pts + (size_t)i * 8;
    float enh[10];
    enh[0] = p[1]; enh[1] = p[2]; enh[2] = p[3];
    enh[3] = p[4]; enh[4] = p[5]; enh[5] = p[6]; enh[6] = p[7];
    float v = p[6];
    enh[7] = fabsf(v); enh[8] = v * v;
    enh[9] = (v > 0.f) ? 1.f : ((v < 0.f) ? -1.f : 0.f);
    float acc5 = b2[5];
    for (int o = 0; o < 64; ++o) {
        float h = b1[o];
        #pragma unroll
        for (int k = 0; k < 10; ++k) h += enh[k] * W1[o * 10 + k];
        h = fmaxf(h * g1[o] + be1[o], 0.f);
        acc5 += h * W2[5 * 64 + o];
    }
    float w5 = 1.f / (1.f + expf(-acc5));
    xyz[i * 3 + 0] = enh[0]; xyz[i * 3 + 1] = enh[1]; xyz[i * 3 + 2] = enh[2];
    float* xo = x0 + (size_t)i * 11;
    #pragma unroll
    for (int k = 0; k < 10; ++k) xo[k] = enh[k];
    xo[10] = w5 * enh[5];
}

// ---------------------------------------------------------------------------
// Farthest point sampling: single workgroup, wave32 shuffle argmax reduce
// ---------------------------------------------------------------------------
__global__ void k_fps(const float* __restrict__ xyz, int n, int npoint,
                      int* __restrict__ out_idx, float* __restrict__ dist)
{
    __shared__ float s_best[32];
    __shared__ int   s_bi[32];
    __shared__ int   s_far;
    int tid = threadIdx.x;
    int nthreads = blockDim.x;
    int lane = tid & 31, wave = tid >> 5;
    int nwaves = nthreads >> 5;
    for (int i = tid; i < n; i += nthreads) dist[i] = 1e10f;
    if (tid == 0) s_far = 0;
    __syncthreads();
    for (int it = 0; it < npoint; ++it) {
        int far = s_far;
        if (tid == 0) out_idx[it] = far;
        float fx = xyz[far * 3 + 0], fy = xyz[far * 3 + 1], fz = xyz[far * 3 + 2];
        float best = -1.f; int bi = 0;
        for (int i = tid; i < n; i += nthreads) {
            float dx = xyz[i * 3 + 0] - fx, dy = xyz[i * 3 + 1] - fy, dz = xyz[i * 3 + 2] - fz;
            float d = dx * dx + dy * dy + dz * dz;
            float dd = fminf(dist[i], d);
            dist[i] = dd;
            if (dd > best) { best = dd; bi = i; }
        }
        for (int off = 16; off > 0; off >>= 1) {
            float ob = __shfl_down(best, off, 32);
            int   oi = __shfl_down(bi, off, 32);
            if (ob > best) { best = ob; bi = oi; }
        }
        if (lane == 0) { s_best[wave] = best; s_bi[wave] = bi; }
        __syncthreads();
        if (wave == 0) {
            float bb = (lane < nwaves) ? s_best[lane] : -2.f;
            int   ii = (lane < nwaves) ? s_bi[lane] : 0;
            for (int off = 16; off > 0; off >>= 1) {
                float ob = __shfl_down(bb, off, 32);
                int   oi = __shfl_down(ii, off, 32);
                if (ob > bb) { bb = ob; ii = oi; }
            }
            if (lane == 0) s_far = ii;
        }
        __syncthreads();
    }
}

__global__ void k_gather_xyz(const float* __restrict__ xyz, const int* __restrict__ idx,
                             int m, float* __restrict__ out)
{
    int t = blockIdx.x * blockDim.x + threadIdx.x;
    if (t >= m * 3) return;
    int q = t / 3, c = t % 3;
    out[t] = xyz[idx[q] * 3 + c];
}

// Ball query: first k indices (in increasing order) within radius; pad with first
__global__ void k_ballq(const float* __restrict__ xyz, const float* __restrict__ nxyz,
                        int n, int m, float r2, int k, int* __restrict__ gidx)
{
    int q = blockIdx.x * blockDim.x + threadIdx.x;
    if (q >= m) return;
    float qx = nxyz[q * 3], qy = nxyz[q * 3 + 1], qz = nxyz[q * 3 + 2];
    int cnt = 0;
    int* out = gidx + (size_t)q * k;
    for (int i = 0; i < n && cnt < k; ++i) {
        float dx = xyz[i * 3] - qx, dy = xyz[i * 3 + 1] - qy, dz = xyz[i * 3 + 2] - qz;
        if (dx * dx + dy * dy + dz * dz <= r2) out[cnt++] = i;
    }
    int first = (cnt > 0) ? out[0] : 0;
    for (; cnt < k; ++cnt) out[cnt] = first;
}

// Grouped rows: [xyz[gi]-new_xyz[q], feat[gi], 0-pad to Kp]
__global__ void k_group(const float* __restrict__ xyz, const float* __restrict__ feat, int C,
                        const float* __restrict__ nxyz, const int* __restrict__ gidx,
                        int k, int m, int Kp, float* __restrict__ X)
{
    int r = blockIdx.x * blockDim.x + threadIdx.x;
    if (r >= m * k) return;
    int q = r / k;
    int gi = gidx[r];
    float* row = X + (size_t)r * Kp;
    row[0] = xyz[gi * 3 + 0] - nxyz[q * 3 + 0];
    row[1] = xyz[gi * 3 + 1] - nxyz[q * 3 + 1];
    row[2] = xyz[gi * 3 + 2] - nxyz[q * 3 + 2];
    const float* f = feat + (size_t)gi * C;
    for (int c = 0; c < C; ++c) row[3 + c] = f[c];
    for (int c = 3 + C; c < Kp; ++c) row[c] = 0.f;
}

__global__ void k_maxpool(const float* __restrict__ Yin, int m, int k, int C,
                          float* __restrict__ out)
{
    int t = blockIdx.x * blockDim.x + threadIdx.x;
    if (t >= m * C) return;
    int q = t / C, c = t % C;
    float mx = -3.4e38f;
    for (int j = 0; j < k; ++j) mx = fmaxf(mx, Yin[(size_t)(q * k + j) * C + c]);
    out[t] = mx;
}

__global__ void k_colmax(const float* __restrict__ Yin, int m, int C, float* __restrict__ out)
{
    int c = blockIdx.x * blockDim.x + threadIdx.x;
    if (c >= C) return;
    float mx = -3.4e38f;
    for (int r = 0; r < m; ++r) mx = fmaxf(mx, Yin[(size_t)r * C + c]);
    out[c] = mx;
}

// 3-NN with inverse-distance weights
__global__ void k_knn3(const float* __restrict__ q_xyz, int nq,
                       const float* __restrict__ s_xyz, int ns,
                       int* __restrict__ idx3, float* __restrict__ w3)
{
    int q = blockIdx.x * blockDim.x + threadIdx.x;
    if (q >= nq) return;
    float qx = q_xyz[q * 3], qy = q_xyz[q * 3 + 1], qz = q_xyz[q * 3 + 2];
    float b0 = 1e30f, b1 = 1e30f, b2 = 1e30f; int i0 = 0, i1 = 0, i2 = 0;
    for (int i = 0; i < ns; ++i) {
        float dx = s_xyz[i * 3] - qx, dy = s_xyz[i * 3 + 1] - qy, dz = s_xyz[i * 3 + 2] - qz;
        float d = dx * dx + dy * dy + dz * dz;
        if (d < b0)      { b2 = b1; i2 = i1; b1 = b0; i1 = i0; b0 = d; i0 = i; }
        else if (d < b1) { b2 = b1; i2 = i1; b1 = d;  i1 = i; }
        else if (d < b2) { b2 = d;  i2 = i; }
    }
    float w0 = 1.f / (b0 + 1e-8f), w1 = 1.f / (b1 + 1e-8f), w2 = 1.f / (b2 + 1e-8f);
    float s = w0 + w1 + w2;
    idx3[q * 3 + 0] = i0; idx3[q * 3 + 1] = i1; idx3[q * 3 + 2] = i2;
    w3[q * 3 + 0] = w0 / s; w3[q * 3 + 1] = w1 / s; w3[q * 3 + 2] = w2 / s;
}

// X[q] = [A[q](CA), sum_j w_j*B[idx_j](CB), 0-pad to Kp]
__global__ void k_concat_interp(const float* __restrict__ A, int CA,
                                const float* __restrict__ B, int CB,
                                const int* __restrict__ idx3, const float* __restrict__ w3,
                                int nq, int Kp, float* __restrict__ X)
{
    int t = blockIdx.x * blockDim.x + threadIdx.x;
    if (t >= nq * Kp) return;
    int q = t / Kp, c = t % Kp;
    float v = 0.f;
    if (c < CA) v = A[(size_t)q * CA + c];
    else if (c < CA + CB) {
        int cc = c - CA;
        const int* id = idx3 + q * 3;
        const float* w = w3 + q * 3;
        v = w[0] * B[(size_t)id[0] * CB + cc] + w[1] * B[(size_t)id[1] * CB + cc]
          + w[2] * B[(size_t)id[2] * CB + cc];
    }
    X[t] = v;
}

// X[q] = [A[q](CA), B[q](CB), 0-pad to Kp]
__global__ void k_concat(const float* __restrict__ A, int CA, const float* __restrict__ B, int CB,
                         int nq, int Kp, float* __restrict__ X)
{
    int t = blockIdx.x * blockDim.x + threadIdx.x;
    if (t >= nq * Kp) return;
    int q = t / Kp, c = t % Kp;
    float v = 0.f;
    if (c < CA) v = A[(size_t)q * CA + c];
    else if (c < CA + CB) v = B[(size_t)q * CB + (c - CA)];
    X[t] = v;
}

// X[q] = [A[q](CA), brow(CB), 0-pad to Kp]
__global__ void k_concat_bcast(const float* __restrict__ A, int CA,
                               const float* __restrict__ brow, int CB,
                               int nq, int Kp, float* __restrict__ X)
{
    int t = blockIdx.x * blockDim.x + threadIdx.x;
    if (t >= nq * Kp) return;
    int q = t / Kp, c = t % Kp;
    float v = 0.f;
    if (c < CA) v = A[(size_t)q * CA + c];
    else if (c < CA + CB) v = brow[c - CA];
    X[t] = v;
}

// Classifier head: 64 -> 32 (affine+relu) -> 1 sigmoid
__global__ void k_cls(const float* __restrict__ l0f,
                      const float* __restrict__ W1, const float* __restrict__ b1,
                      const float* __restrict__ g1, const float* __restrict__ be1,
                      const float* __restrict__ W2, const float* __restrict__ b2,
                      float* __restrict__ mos)
{
    int i = blockIdx.x * blockDim.x + threadIdx.x;
    if (i >= NPTS) return;
    const float* x = l0f + (size_t)i * 64;
    float acc = b2[0];
    for (int o = 0; o < 32; ++o) {
        float h = b1[o];
        for (int k = 0; k < 64; ++k) h += x[k] * W1[o * 64 + k];
        h = fmaxf(h * g1[o] + be1[o], 0.f);
        acc += h * W2[o];
    }
    mos[i] = 1.f / (1.f + expf(-acc));
}

// ---------------------------------------------------------------------------
extern "C" void kernel_launch(void* const* d_in, const int* in_sizes, int n_in,
                              void* d_out, int out_size, void* d_ws, size_t ws_size,
                              hipStream_t stream)
{
    (void)in_sizes; (void)n_in; (void)out_size; (void)ws_size;
    int ai = 0;
    auto nx = [&]() { return (const float*)d_in[ai++]; };
    const float* points = nx();
    const float *vaW1 = nx(), *vab1 = nx(), *vag1 = nx(), *vabe1 = nx(), *vaW2 = nx(), *vab2 = nx();
    struct L { const float *W, *b, *g, *be; int cin, cout, Kp; };
    auto Lget = [&](int cin, int cout) {
        L l; l.W = nx(); l.b = nx(); l.g = nx(); l.be = nx();
        l.cin = cin; l.cout = cout; l.Kp = (cin + 31) & ~31; return l;
    };
    L Ls[16];
    Ls[0]  = Lget(14, 64);    Ls[1]  = Lget(64, 64);    Ls[2]  = Lget(64, 128);    // sa1
    Ls[3]  = Lget(131, 128);  Ls[4]  = Lget(128, 128);  Ls[5]  = Lget(128, 256);   // sa2
    Ls[6]  = Lget(259, 256);  Ls[7]  = Lget(256, 512);  Ls[8]  = Lget(512, 1024);  // sa3
    Ls[9]  = Lget(1280, 256); Ls[10] = Lget(256, 256);                             // fp3
    Ls[11] = Lget(384, 256);  Ls[12] = Lget(256, 128);                             // fp2
    Ls[13] = Lget(139, 128);  Ls[14] = Lget(128, 128);  Ls[15] = Lget(128, 64);    // fp1
    const float *cW1 = nx(), *cb1 = nx(), *cg1 = nx(), *cbe1 = nx(), *cW2 = nx(), *cb2 = nx();

    // workspace carving (float units)
    float* base = (float*)d_ws;
    size_t off = 0;
    auto alloc = [&](size_t n) { float* p = base + off; off += (n + 15) & ~(size_t)15; return p; };
    float* xyz   = alloc((size_t)NPTS * 3);
    float* x0    = alloc((size_t)NPTS * 11);
    float* fdist = alloc(NPTS);
    int*   l1idx = (int*)alloc(1024);
    int*   l2idx = (int*)alloc(256);
    float* l1xyz = alloc(1024 * 3);
    float* l2xyz = alloc(256 * 3);
    int*   gidx1 = (int*)alloc(1024 * 32);
    int*   gidx2 = (int*)alloc(256 * 64);
    float* Xbuf  = alloc((size_t)NPTS * 160);
    float* actP  = alloc((size_t)NPTS * 128);
    float* actQ  = alloc((size_t)NPTS * 128);
    float* l2buf = alloc(256 * 256);
    float* l2f   = alloc(256 * 256);
    float* l3    = alloc(1024);
    int*   idx3a = (int*)alloc(1024 * 3);
    float* w3a   = alloc(1024 * 3);
    int*   idx3b = (int*)alloc((size_t)NPTS * 3);
    float* w3b   = alloc((size_t)NPTS * 3);
    _Float16* Wh[16];
    for (int i = 0; i < 16; ++i) {
        size_t n = (size_t)Ls[i].Kp * Ls[i].cout;           // packed size uses Kp
        Wh[i] = (_Float16*)alloc((n + 1) / 2);
    }
    float* mos = (float*)d_out;
    float* l1  = (float*)d_out + NPTS;                 // (1024,128)
    float* l1f = (float*)d_out + NPTS + 1024 * 128;    // (1024,128)

    // pack weights f32 -> fragment-ordered f16
    for (int i = 0; i < 16; ++i) {
        int n = Ls[i].cout * Ls[i].Kp;
        k_pack<<<(n + 255) / 256, 256, 0, stream>>>(Ls[i].W, Wh[i], Ls[i].cin, Ls[i].Kp, Ls[i].cout);
    }
    auto gemm = [&](const float* X, int li, float* Y, int M) {
        dim3 g(M / 64, Ls[li].cout / 64);
        k_gemm<<<g, 128, 0, stream>>>(X, (const v16h*)Wh[li], Ls[li].b, Ls[li].g, Ls[li].be, Y,
                                      M, Ls[li].Kp, Ls[li].cout);
    };

    // preprocess
    k_preprocess<<<(NPTS + 255) / 256, 256, 0, stream>>>(points, vaW1, vab1, vag1, vabe1,
                                                         vaW2, vab2, xyz, x0);
    // SA1
    k_fps<<<1, 1024, 0, stream>>>(xyz, NPTS, 1024, l1idx, fdist);
    k_gather_xyz<<<(1024 * 3 + 255) / 256, 256, 0, stream>>>(xyz, l1idx, 1024, l1xyz);
    k_ballq<<<4, 256, 0, stream>>>(xyz, l1xyz, NPTS, 1024, 0.04f, 32, gidx1);
    k_group<<<(1024 * 32 + 255) / 256, 256, 0, stream>>>(xyz, x0, 11, l1xyz, gidx1, 32, 1024,
                                                         Ls[0].Kp, Xbuf);
    gemm(Xbuf, 0, actP, 32768);
    gemm(actP, 1, actQ, 32768);
    gemm(actQ, 2, actP, 32768);
    k_maxpool<<<(1024 * 128 + 255) / 256, 256, 0, stream>>>(actP, 1024, 32, 128, l1);

    // SA2
    k_fps<<<1, 1024, 0, stream>>>(l1xyz, 1024, 256, l2idx, fdist);
    k_gather_xyz<<<(256 * 3 + 255) / 256, 256, 0, stream>>>(l1xyz, l2idx, 256, l2xyz);
    k_ballq<<<1, 256, 0, stream>>>(l1xyz, l2xyz, 1024, 256, 0.16f, 64, gidx2);
    k_group<<<(256 * 64 + 255) / 256, 256, 0, stream>>>(l1xyz, l1, 128, l2xyz, gidx2, 64, 256,
                                                        Ls[3].Kp, Xbuf);
    gemm(Xbuf, 3, actP, 16384);
    gemm(actP, 4, actQ, 16384);
    gemm(actQ, 5, actP, 16384);
    k_maxpool<<<(256 * 256 + 255) / 256, 256, 0, stream>>>(actP, 256, 64, 256, l2buf);

    // sa3 global MLP + global max
    k_concat<<<(256 * Ls[6].Kp + 255) / 256, 256, 0, stream>>>(l2xyz, 3, l2buf, 256, 256,
                                                               Ls[6].Kp, Xbuf);
    gemm(Xbuf, 6, actP, 256);
    gemm(actP, 7, actQ, 256);
    gemm(actQ, 8, actP, 256);
    k_colmax<<<4, 256, 0, stream>>>(actP, 256, 1024, l3);

    // fp3 (broadcast interp)
    k_concat_bcast<<<(256 * 1280 + 255) / 256, 256, 0, stream>>>(l2buf, 256, l3, 1024, 256,
                                                                 Ls[9].Kp, Xbuf);
    gemm(Xbuf, 9, actP, 256);
    gemm(actP, 10, l2f, 256);

    // fp2
    k_knn3<<<4, 256, 0, stream>>>(l1xyz, 1024, l2xyz, 256, idx3a, w3a);
    k_concat_interp<<<(1024 * 384 + 255) / 256, 256, 0, stream>>>(l1, 128, l2f, 256,
                                                                  idx3a, w3a, 1024, Ls[11].Kp, Xbuf);
    gemm(Xbuf, 11, actP, 1024);
    gemm(actP, 12, l1f, 1024);

    // fp1
    k_knn3<<<(NPTS + 255) / 256, 256, 0, stream>>>(xyz, NPTS, l1xyz, 1024, idx3b, w3b);
    k_concat_interp<<<((NPTS * Ls[13].Kp) + 255) / 256, 256, 0, stream>>>(x0, 11, l1f, 128,
                                                                          idx3b, w3b, NPTS,
                                                                          Ls[13].Kp, Xbuf);
    gemm(Xbuf, 13, actP, 32768);
    gemm(actP, 14, actQ, 32768);
    gemm(actQ, 15, actP, 32768);

    // classifier -> mos
    k_cls<<<(NPTS + 255) / 256, 256, 0, stream>>>(actP, cW1, cb1, cg1, cbe1, cW2, cb2, mos);
}
